// LLaMANet_35519379538451
// MI455X (gfx1250) — matrix operand order repaired and checked by reading, not compiled, not run on hardware
//
#include <hip/hip_runtime.h>

// ---------------- model constants ----------------
#define B_    4
#define S_    1024
#define D_    2048
#define NL_   2
#define H_    16
#define KV_   4
#define F_    8192
#define R_    16
#define HD_   128
#define KVD_  (KV_ * HD_)        // 512
#define NTOK_ (B_ * S_)          // 4096
#define LORA_SCALE_ 2.0f
#define ROPE_THETA_ 500000.0f
#define EPS_ 1e-5f

typedef __attribute__((ext_vector_type(16))) __bf16 v16bf;
typedef __attribute__((ext_vector_type(8)))  float  v8f;

union AFrag { v16bf v; unsigned u[8]; };
union CTile { v8f   v; float    f[8]; };

// ---------------- helpers ----------------
__device__ __forceinline__ float bf2f(unsigned short h) {
    unsigned x = ((unsigned)h) << 16;
    return __builtin_bit_cast(float, x);
}
__device__ __forceinline__ unsigned short f2bf(float f) {
    unsigned x = __builtin_bit_cast(unsigned, f);
    x += 0x7FFFu + ((x >> 16) & 1u);          // round to nearest even
    return (unsigned short)(x >> 16);
}
__device__ __forceinline__ float half16_max(float v) {
    #pragma unroll
    for (int m = 1; m < 16; m <<= 1) v = fmaxf(v, __shfl_xor(v, m, 32));
    return v;
}
__device__ __forceinline__ float half16_sum(float v) {
    #pragma unroll
    for (int m = 1; m < 16; m <<= 1) v += __shfl_xor(v, m, 32);
    return v;
}
// pair base K index for 16-bit A-matrix fragment, vgpr j, half (0/1)
__device__ __forceinline__ int a_pair(int j, int hi) {
    return 2 * j + hi * 16;
}

// ---------------- elementwise kernels ----------------
__global__ void embed_kernel(const float* __restrict__ embed,
                             const int* __restrict__ ids,
                             float* __restrict__ x) {
    int idx = blockIdx.x * blockDim.x + threadIdx.x;     // one per 4 floats
    int m = idx / (D_ / 4);
    int c = idx % (D_ / 4);
    const float4* src = (const float4*)(embed + (size_t)ids[m] * D_) + c;
    float4* dst = (float4*)(x + (size_t)m * D_) + c;
    *dst = *src;
}

__global__ void f32_to_bf16_kernel(const float* __restrict__ src,
                                   unsigned short* __restrict__ dst, int n) {
    int i = blockIdx.x * blockDim.x + threadIdx.x;
    if (i < n) dst[i] = f2bf(src[i]);
}

__global__ __launch_bounds__(256)
void rmsnorm_bf16_kernel(const float* __restrict__ x,
                         const float* __restrict__ w,
                         unsigned short* __restrict__ out) {
    __shared__ float red[256];
    int row = blockIdx.x, tid = threadIdx.x;
    const float* xr = x + (size_t)row * D_;
    float ss = 0.f;
    #pragma unroll
    for (int i = 0; i < D_ / 256; i++) { float v = xr[tid + i * 256]; ss += v * v; }
    red[tid] = ss; __syncthreads();
    for (int s = 128; s > 0; s >>= 1) { if (tid < s) red[tid] += red[tid + s]; __syncthreads(); }
    float inv = rsqrtf(red[0] / (float)D_ + EPS_);
    unsigned short* orow = out + (size_t)row * D_;
    #pragma unroll
    for (int i = 0; i < D_ / 256; i++) {
        int c = tid + i * 256;
        orow[c] = f2bf(xr[c] * inv * w[c]);
    }
}

__global__ __launch_bounds__(256)
void rmsnorm_f32_kernel(const float* __restrict__ x,
                        const float* __restrict__ w,
                        float* __restrict__ out) {
    __shared__ float red[256];
    int row = blockIdx.x, tid = threadIdx.x;
    const float* xr = x + (size_t)row * D_;
    float ss = 0.f;
    #pragma unroll
    for (int i = 0; i < D_ / 256; i++) { float v = xr[tid + i * 256]; ss += v * v; }
    red[tid] = ss; __syncthreads();
    for (int s = 128; s > 0; s >>= 1) { if (tid < s) red[tid] += red[tid + s]; __syncthreads(); }
    float inv = rsqrtf(red[0] / (float)D_ + EPS_);
    float* orow = out + (size_t)row * D_;
    #pragma unroll
    for (int i = 0; i < D_ / 256; i++) {
        int c = tid + i * 256;
        orow[c] = xr[c] * inv * w[c];
    }
}

// LoRA down: t[m,r] = sum_k h[m,k] * la[r,k]   (h bf16, la f32 row-major [R,K])
__global__ void lora_down_kernel(const unsigned short* __restrict__ h,
                                 const float* __restrict__ la,
                                 float* __restrict__ t) {
    int idx = blockIdx.x * blockDim.x + threadIdx.x;  // m*R + r
    int r = idx & (R_ - 1); int m = idx >> 4;
    const unsigned short* hp = h + (size_t)m * D_;
    const float* lp = la + (size_t)r * D_;
    float s = 0.f;
    for (int k = 0; k < D_; k++) s += bf2f(hp[k]) * lp[k];
    t[idx] = s;
}

// LoRA up + scaled add: q[m,n] += 2 * sum_r t[m,r] * lb[n,r]   (lb row-major [N,R])
__global__ void lora_up_add_kernel(const float* __restrict__ t,
                                   const float* __restrict__ lb,
                                   float* __restrict__ q, int N) {
    int idx = blockIdx.x * blockDim.x + threadIdx.x;  // m*N + n
    int n = idx % N; int m = idx / N;
    const float* tp = t + (size_t)m * R_;
    const float* lp = lb + (size_t)n * R_;
    float s = 0.f;
    #pragma unroll
    for (int r = 0; r < R_; r++) s += tp[r] * lp[r];
    q[idx] += LORA_SCALE_ * s;
}

// RoPE (LLaMA rotate_half) f32 in -> bf16 out; layout [tok, nh, HD]
__global__ void rope_kernel(const float* __restrict__ in,
                            unsigned short* __restrict__ out, int nh) {
    int idx = blockIdx.x * blockDim.x + threadIdx.x;   // tok*nh*64
    int j  = idx & 63;
    int hh = (idx >> 6) % nh;
    int m  = idx / (64 * nh);
    int s  = m & (S_ - 1);
    float inv = __expf(-__logf(ROPE_THETA_) * ((2.0f * (float)j) / (float)HD_));
    float ang = (float)s * inv;
    float c = cosf(ang), sn = sinf(ang);
    size_t base = ((size_t)m * nh + hh) * HD_;
    float x0 = in[base + j], x1 = in[base + j + 64];
    out[base + j]      = f2bf(x0 * c - x1 * sn);
    out[base + j + 64] = f2bf(x1 * c + x0 * sn);
}

// masked mean pool over sequence
__global__ void pool_kernel(const float* __restrict__ hf,
                            const int* __restrict__ mask,
                            float* __restrict__ out) {
    int idx = blockIdx.x * blockDim.x + threadIdx.x;   // b*D + d
    int d = idx % D_; int b = idx / D_;
    float s = 0.f, den = 0.f;
    for (int t = 0; t < S_; t++) {
        float mm = (float)mask[b * S_ + t];
        s += hf[((size_t)(b * S_ + t)) * D_ + d] * mm;
        den += mm;
    }
    out[idx] = s / fmaxf(den, 1e-9f);
}

// ---------------- WMMA GEMM: C[M,N] (+)= A_bf16[M,K] @ B_bf16[K,N] ----------------
// block = 256 thr (8 waves); block tile 128x64; wave tile 16x64 (4 C tiles)
__global__ __launch_bounds__(256)
void gemm_bf16_kernel(const unsigned short* __restrict__ A,
                      const unsigned short* __restrict__ B,
                      float* __restrict__ C,
                      int M, int N, int K, int accumulate) {
    __shared__ unsigned short Bs[64 * 32];   // transposed: Bs[n][k]
    const int tid  = threadIdx.x;
    const int lane = tid & 31;
    const int wave = tid >> 5;
    const int hi   = (lane >> 4) & 1;
    const int l15  = lane & 15;
    const int nb   = blockIdx.x * 64;
    const int mbase = blockIdx.y * 128 + wave * 16;
    const int arow  = mbase + l15;
    // staging map: thread covers K-pair (2kp,2kp+1) x 4 columns
    const int kp = tid >> 4;          // 0..15
    const int nn = (tid & 15) * 4;    // 0..60

    CTile acc[4];
    #pragma unroll
    for (int t = 0; t < 4; t++)
        #pragma unroll
        for (int r = 0; r < 8; r++) acc[t].f[r] = 0.f;

    for (int k0 = 0; k0 < K; k0 += 32) {
        // stage 32x64 B tile, transposed, packed 32-bit LDS stores
        {
            const unsigned* s0 = (const unsigned*)(B + (size_t)(k0 + 2 * kp) * N + nb + nn);
            const unsigned* s1 = (const unsigned*)(B + (size_t)(k0 + 2 * kp + 1) * N + nb + nn);
            unsigned a0 = s0[0], a1 = s0[1];
            unsigned b0 = s1[0], b1 = s1[1];
            unsigned* Bs32 = (unsigned*)Bs;    // word (n*32 + 2kp)/2 = n*16 + kp
            Bs32[(nn + 0) * 16 + kp] = (a0 & 0xFFFFu) | (b0 << 16);
            Bs32[(nn + 1) * 16 + kp] = (a0 >> 16)     | (b0 & 0xFFFF0000u);
            Bs32[(nn + 2) * 16 + kp] = (a1 & 0xFFFFu) | (b1 << 16);
            Bs32[(nn + 3) * 16 + kp] = (a1 >> 16)     | (b1 & 0xFFFF0000u);
        }
        __syncthreads();
        if (k0 + 32 < K)
            __builtin_prefetch(B + (size_t)(k0 + 32 + 2 * kp) * N + nb + nn, 0, 1);
        // A fragment (16x32, lane=row, K pairs per VGPR)
        AFrag a;
        #pragma unroll
        for (int j = 0; j < 8; j++) {
            int kk = a_pair(j, hi);
            a.u[j] = *(const unsigned*)(A + (size_t)arow * K + k0 + kk);
        }
        #pragma unroll
        for (int t = 0; t < 4; t++) {
            AFrag bf;
            #pragma unroll
            for (int j = 0; j < 8; j++) {
                int kk = 2 * j + hi * 16;
                bf.u[j] = *(const unsigned*)(&Bs[(t * 16 + l15) * 32 + kk]);
            }
            acc[t].v = __builtin_amdgcn_wmma_f32_16x16x32_bf16(
                false, a.v, false, bf.v, (short)0, acc[t].v, false, false);
        }
        __syncthreads();
    }
    #pragma unroll
    for (int t = 0; t < 4; t++) {
        int col = nb + t * 16 + l15;
        #pragma unroll
        for (int r = 0; r < 8; r++) {
            int row = mbase + r + hi * 8;
            size_t off = (size_t)row * N + col;
            if (accumulate) C[off] += acc[t].f[r]; else C[off] = acc[t].f[r];
        }
    }
}

// ---------------- fused gate/up GEMM: out = silu(A@G) * (A@U), bf16 out ----------------
// block 256 thr; block tile 128x32; wave tile 16x32 (2 C tiles per matrix)
__global__ __launch_bounds__(256)
void gateup_bf16_kernel(const unsigned short* __restrict__ A,
                        const unsigned short* __restrict__ G,
                        const unsigned short* __restrict__ U,
                        unsigned short* __restrict__ out,
                        int M, int N, int K) {
    __shared__ unsigned short Gs[32 * 32];
    __shared__ unsigned short Us[32 * 32];
    const int tid  = threadIdx.x;
    const int lane = tid & 31;
    const int wave = tid >> 5;
    const int hi   = (lane >> 4) & 1;
    const int l15  = lane & 15;
    const int nb   = blockIdx.x * 32;
    const int mbase = blockIdx.y * 128 + wave * 16;
    const int arow  = mbase + l15;
    // staging map: 128 threads per matrix, K-pair x 4 columns each
    const int sel = tid >> 7;          // 0 -> G, 1 -> U
    const int t2  = tid & 127;
    const int kp  = t2 >> 3;           // 0..15
    const int nn  = (t2 & 7) * 4;      // 0..28

    CTile ag[2], au[2];
    #pragma unroll
    for (int t = 0; t < 2; t++)
        #pragma unroll
        for (int r = 0; r < 8; r++) { ag[t].f[r] = 0.f; au[t].f[r] = 0.f; }

    for (int k0 = 0; k0 < K; k0 += 32) {
        {
            const unsigned short* W = sel ? U : G;
            unsigned* Ls32 = (unsigned*)(sel ? Us : Gs);
            const unsigned* s0 = (const unsigned*)(W + (size_t)(k0 + 2 * kp) * N + nb + nn);
            const unsigned* s1 = (const unsigned*)(W + (size_t)(k0 + 2 * kp + 1) * N + nb + nn);
            unsigned a0 = s0[0], a1 = s0[1];
            unsigned b0 = s1[0], b1 = s1[1];
            Ls32[(nn + 0) * 16 + kp] = (a0 & 0xFFFFu) | (b0 << 16);
            Ls32[(nn + 1) * 16 + kp] = (a0 >> 16)     | (b0 & 0xFFFF0000u);
            Ls32[(nn + 2) * 16 + kp] = (a1 & 0xFFFFu) | (b1 << 16);
            Ls32[(nn + 3) * 16 + kp] = (a1 >> 16)     | (b1 & 0xFFFF0000u);
        }
        __syncthreads();
        AFrag a;
        #pragma unroll
        for (int j = 0; j < 8; j++) {
            int kk = a_pair(j, hi);
            a.u[j] = *(const unsigned*)(A + (size_t)arow * K + k0 + kk);
        }
        #pragma unroll
        for (int t = 0; t < 2; t++) {
            AFrag bg, bu;
            #pragma unroll
            for (int j = 0; j < 8; j++) {
                int kk = 2 * j + hi * 16;
                bg.u[j] = *(const unsigned*)(&Gs[(t * 16 + l15) * 32 + kk]);
                bu.u[j] = *(const unsigned*)(&Us[(t * 16 + l15) * 32 + kk]);
            }
            ag[t].v = __builtin_amdgcn_wmma_f32_16x16x32_bf16(
                false, a.v, false, bg.v, (short)0, ag[t].v, false, false);
            au[t].v = __builtin_amdgcn_wmma_f32_16x16x32_bf16(
                false, a.v, false, bu.v, (short)0, au[t].v, false, false);
        }
        __syncthreads();
    }
    #pragma unroll
    for (int t = 0; t < 2; t++) {
        int col = nb + t * 16 + l15;
        #pragma unroll
        for (int r = 0; r < 8; r++) {
            int row = mbase + r + hi * 8;
            float g = ag[t].f[r], u = au[t].f[r];
            float sil = g / (1.0f + __expf(-g));
            out[(size_t)row * N + col] = f2bf(sil * u);
        }
    }
}

// ---------------- flash attention (one wave per 16-query tile per (b,h)) ----------------
__global__ __launch_bounds__(32)
void attn_kernel(const unsigned short* __restrict__ qb,
                 const unsigned short* __restrict__ kb,
                 const unsigned short* __restrict__ vb,
                 const int* __restrict__ amask,
                 unsigned short* __restrict__ ob) {
    __shared__ unsigned short Vs[32 * 128];   // V tile, row-major [k][hd], async-filled
    __shared__ unsigned short plds[16 * 32];  // P tile for layout change
    const int lane = threadIdx.x;
    const int hi   = lane >> 4;
    const int l15  = lane & 15;
    const int qt = blockIdx.x;
    const int bh = blockIdx.y;
    const int b  = bh / H_, hh = bh % H_;
    const int kvh = hh / (H_ / KV_);
    const unsigned short* qp = qb + (size_t)(b * S_) * D_   + (size_t)hh  * HD_;
    const unsigned short* kp = kb + (size_t)(b * S_) * KVD_ + (size_t)kvh * HD_;
    const unsigned short* vp = vb + (size_t)(b * S_) * KVD_ + (size_t)kvh * HD_;
    const float scale = 0.088388347648318447f;  // 1/sqrt(128)
    const unsigned vs_base = (unsigned)(size_t)(&Vs[0]);

    // Q fragments: 16 rows x 128 (4 x 16x32 A-frags)
    AFrag qa[4];
    #pragma unroll
    for (int f = 0; f < 4; f++)
        #pragma unroll
        for (int j = 0; j < 8; j++) {
            int kk = a_pair(j, hi);
            qa[f].u[j] = *(const unsigned*)(qp + (size_t)(qt * 16 + l15) * D_ + f * 32 + kk);
        }

    CTile accO[8];
    #pragma unroll
    for (int nt = 0; nt < 8; nt++)
        #pragma unroll
        for (int r = 0; r < 8; r++) accO[nt].f[r] = 0.f;
    float mr[8], lr[8];
    #pragma unroll
    for (int r = 0; r < 8; r++) { mr[r] = -1e30f; lr[r] = 0.f; }

    const int kend = qt * 16 + 16;     // causal bound
    for (int k32 = 0; k32 < kend; k32 += 32) {
        // async-stage the 32x128 V tile into LDS (overlaps with Q.K^T below).
        // s_wait_dscnt 0 first: previous iteration's LDS reads of Vs are done.
        asm volatile("s_wait_dscnt 0x0" ::: "memory");
        #pragma unroll
        for (int i = 0; i < 16; i++) {
            int chunk = i * 32 + lane;           // 512 x 16B chunks
            int kk   = chunk >> 4;               // key row 0..31
            int part = chunk & 15;               // 16B piece within 256B row
            unsigned loff = vs_base + (unsigned)(kk * 256 + part * 16);
            unsigned long long ga = (unsigned long long)(size_t)
                ((const char*)(vp + (size_t)(k32 + kk) * KVD_) + part * 16);
            asm volatile("global_load_async_to_lds_b128 %0, %1, off"
                         :: "v"(loff), "v"(ga) : "memory");
        }

        // scores: two 16x16 tiles, K loop over 128
        CTile sc[2];
        #pragma unroll
        for (int t = 0; t < 2; t++) {
            #pragma unroll
            for (int r = 0; r < 8; r++) sc[t].f[r] = 0.f;
            #pragma unroll
            for (int f = 0; f < 4; f++) {
                AFrag kf;
                #pragma unroll
                for (int j = 0; j < 8; j++) {
                    int kk = 2 * j + hi * 16;
                    kf.u[j] = *(const unsigned*)(kp + (size_t)(k32 + t * 16 + l15) * KVD_ + f * 32 + kk);
                }
                sc[t].v = __builtin_amdgcn_wmma_f32_16x16x32_bf16(
                    false, qa[f].v, false, kf.v, (short)0, sc[t].v, false, false);
            }
        }
        // online softmax
        #pragma unroll
        for (int r = 0; r < 8; r++) {
            int m  = r + hi * 8;
            int qg = qt * 16 + m;
            int kg0 = k32 + l15, kg1 = k32 + 16 + l15;
            float s0 = sc[0].f[r] * scale;
            float s1 = sc[1].f[r] * scale;
            if (kg0 > qg || amask[b * S_ + kg0] == 0) s0 = -1e30f;
            if (kg1 > qg || amask[b * S_ + kg1] == 0) s1 = -1e30f;
            float tm = half16_max(fmaxf(s0, s1));
            float nm = fmaxf(mr[r], tm);
            float p0 = (s0 <= -1e29f) ? 0.f : __expf(s0 - nm);
            float p1 = (s1 <= -1e29f) ? 0.f : __expf(s1 - nm);
            float ps = half16_sum(p0 + p1);
            float corr = __expf(mr[r] - nm);
            lr[r] = lr[r] * corr + ps;
            mr[r] = nm;
            #pragma unroll
            for (int nt = 0; nt < 8; nt++) accO[nt].f[r] *= corr;
            plds[m * 32 + l15]      = f2bf(p0);
            plds[m * 32 + 16 + l15] = f2bf(p1);
        }
        __syncthreads();
        // P (16x32) as A fragment
        AFrag pa;
        #pragma unroll
        for (int j = 0; j < 8; j++) {
            int kk = a_pair(j, hi);
            pa.u[j] = *(const unsigned*)(&plds[l15 * 32 + kk]);
        }
        // wait for the async V tile, then O += P @ V  (V 32x128 as 8 B-frags from LDS)
        asm volatile("s_wait_asynccnt 0x0" ::: "memory");
        #pragma unroll
        for (int nt = 0; nt < 8; nt++) {
            AFrag vf;
            #pragma unroll
            for (int j = 0; j < 8; j++) {
                int kk = 2 * j + hi * 16;
                int col = nt * 16 + l15;
                unsigned lo = Vs[kk * 128 + col];
                unsigned hv = Vs[(kk + 1) * 128 + col];
                vf.u[j] = lo | (hv << 16);
            }
            accO[nt].v = __builtin_amdgcn_wmma_f32_16x16x32_bf16(
                false, pa.v, false, vf.v, (short)0, accO[nt].v, false, false);
        }
        __syncthreads();
    }
    // normalize and write bf16 output [b,s,h,hd]
    #pragma unroll
    for (int r = 0; r < 8; r++) {
        float inv = 1.f / fmaxf(lr[r], 1e-20f);
        int m = r + hi * 8;
        size_t row = (size_t)(b * S_ + qt * 16 + m) * D_ + (size_t)hh * HD_;
        #pragma unroll
        for (int nt = 0; nt < 8; nt++)
            ob[row + nt * 16 + l15] = f2bf(accO[nt].f[r] * inv);
    }
}

// ---------------- host side ----------------
extern "C" void kernel_launch(void* const* d_in, const int* in_sizes, int n_in,
                              void* d_out, int out_size, void* d_ws, size_t ws_size,
                              hipStream_t stream) {
    (void)in_sizes; (void)n_in; (void)out_size; (void)ws_size;
    const float* embed = (const float*)d_in[0];
    const float* wq    = (const float*)d_in[1];
    const float* wk    = (const float*)d_in[2];
    const float* wv    = (const float*)d_in[3];
    const float* wo    = (const float*)d_in[4];
    const float* laq   = (const float*)d_in[5];
    const float* lbq   = (const float*)d_in[6];
    const float* lav   = (const float*)d_in[7];
    const float* lbv   = (const float*)d_in[8];
    const float* wg    = (const float*)d_in[9];
    const float* wu    = (const float*)d_in[10];
    const float* wd    = (const float*)d_in[11];
    const float* n1    = (const float*)d_in[12];
    const float* n2    = (const float*)d_in[13];
    const float* nf    = (const float*)d_in[14];
    const int* ids     = (const int*)d_in[15];
    const int* amask   = (const int*)d_in[16];
    float* out = (float*)d_out;

    char* w = (char*)d_ws;
    auto take = [&](size_t bytes) { char* p = w; w += (bytes + 255) & ~size_t(255); return p; };
    float*          x    = (float*)         take((size_t)NTOK_ * D_   * 4);
    unsigned short* hbf  = (unsigned short*)take((size_t)NTOK_ * D_   * 2);
    float*          qf   = (float*)         take((size_t)NTOK_ * D_   * 4);
    float*          kf   = (float*)         take((size_t)NTOK_ * KVD_ * 4);
    float*          vf   = (float*)         take((size_t)NTOK_ * KVD_ * 4);
    float*          tlo  = (float*)         take((size_t)NTOK_ * R_   * 4);
    unsigned short* qbb  = (unsigned short*)take((size_t)NTOK_ * D_   * 2);
    unsigned short* kbb  = (unsigned short*)take((size_t)NTOK_ * KVD_ * 2);
    unsigned short* vbb  = (unsigned short*)take((size_t)NTOK_ * KVD_ * 2);
    unsigned short* obb  = (unsigned short*)take((size_t)NTOK_ * D_   * 2);
    unsigned short* gub  = (unsigned short*)take((size_t)NTOK_ * F_   * 2);
    unsigned short* wb0  = (unsigned short*)take((size_t)D_ * F_ * 2);
    unsigned short* wb1  = (unsigned short*)take((size_t)D_ * F_ * 2);
    float*          hf   = (float*)         take((size_t)NTOK_ * D_   * 4);

    auto cvt = [&](const float* src, unsigned short* dst, int n) {
        f32_to_bf16_kernel<<<(n + 255) / 256, 256, 0, stream>>>(src, dst, n);
    };

    // embedding gather
    embed_kernel<<<(NTOK_ * D_ / 4 + 255) / 256, 256, 0, stream>>>(embed, ids, x);

    for (int l = 0; l < NL_; l++) {
        const float* wq_l  = wq  + (size_t)l * D_ * D_;
        const float* wk_l  = wk  + (size_t)l * D_ * KVD_;
        const float* wv_l  = wv  + (size_t)l * D_ * KVD_;
        const float* wo_l  = wo  + (size_t)l * D_ * D_;
        const float* laq_l = laq + (size_t)l * R_ * D_;
        const float* lbq_l = lbq + (size_t)l * D_ * R_;
        const float* lav_l = lav + (size_t)l * R_ * D_;
        const float* lbv_l = lbv + (size_t)l * KVD_ * R_;
        const float* wg_l  = wg  + (size_t)l * D_ * F_;
        const float* wu_l  = wu  + (size_t)l * D_ * F_;
        const float* wd_l  = wd  + (size_t)l * F_ * D_;

        // h = rmsnorm(x, n1) -> bf16
        rmsnorm_bf16_kernel<<<NTOK_, 256, 0, stream>>>(x, n1 + (size_t)l * D_, hbf);

        // q = h @ wq + lora
        cvt(wq_l, wb0, D_ * D_);
        gemm_bf16_kernel<<<dim3(D_ / 64, NTOK_ / 128), 256, 0, stream>>>(
            hbf, wb0, qf, NTOK_, D_, D_, 0);
        lora_down_kernel<<<NTOK_ * R_ / 256, 256, 0, stream>>>(hbf, laq_l, tlo);
        lora_up_add_kernel<<<NTOK_ * D_ / 256, 256, 0, stream>>>(tlo, lbq_l, qf, D_);

        // k = h @ wk ; v = h @ wv + lora
        cvt(wk_l, wb0, D_ * KVD_);
        gemm_bf16_kernel<<<dim3(KVD_ / 64, NTOK_ / 128), 256, 0, stream>>>(
            hbf, wb0, kf, NTOK_, KVD_, D_, 0);
        cvt(wv_l, wb0, D_ * KVD_);
        gemm_bf16_kernel<<<dim3(KVD_ / 64, NTOK_ / 128), 256, 0, stream>>>(
            hbf, wb0, vf, NTOK_, KVD_, D_, 0);
        lora_down_kernel<<<NTOK_ * R_ / 256, 256, 0, stream>>>(hbf, lav_l, tlo);
        lora_up_add_kernel<<<NTOK_ * KVD_ / 256, 256, 0, stream>>>(tlo, lbv_l, vf, KVD_);

        // rope(q), rope(k), v -> bf16
        rope_kernel<<<NTOK_ * H_ * 64 / 256, 256, 0, stream>>>(qf, qbb, H_);
        rope_kernel<<<NTOK_ * KV_ * 64 / 256, 256, 0, stream>>>(kf, kbb, KV_);
        cvt(vf, vbb, NTOK_ * KVD_);

        // attention -> obb (bf16)
        attn_kernel<<<dim3(S_ / 16, B_ * H_), 32, 0, stream>>>(qbb, kbb, vbb, amask, obb);

        // x += o @ wo
        cvt(wo_l, wb0, D_ * D_);
        gemm_bf16_kernel<<<dim3(D_ / 64, NTOK_ / 128), 256, 0, stream>>>(
            obb, wb0, x, NTOK_, D_, D_, 1);

        // MLP: h2 = rmsnorm(x, n2); gu = silu(h2@wg)*(h2@wu); x += gu @ wd
        rmsnorm_bf16_kernel<<<NTOK_, 256, 0, stream>>>(x, n2 + (size_t)l * D_, hbf);
        cvt(wg_l, wb0, D_ * F_);
        cvt(wu_l, wb1, D_ * F_);
        gateup_bf16_kernel<<<dim3(F_ / 32, NTOK_ / 128), 256, 0, stream>>>(
            hbf, wb0, wb1, gub, NTOK_, F_, D_);
        cvt(wd_l, wb0, F_ * D_);
        gemm_bf16_kernel<<<dim3(D_ / 64, NTOK_ / 128), 256, 0, stream>>>(
            gub, wb0, x, NTOK_, D_, F_, 1);
    }

    // final norm + masked mean pool
    rmsnorm_f32_kernel<<<NTOK_, 256, 0, stream>>>(x, nf, hf);
    pool_kernel<<<B_ * D_ / 256, 256, 0, stream>>>(hf, amask, out);
}
